// Net_1013612282379
// MI455X (gfx1250) — matrix operand order repaired and checked by reading, not compiled, not run on hardware
//
#include <hip/hip_runtime.h>
#include <math.h>

#define N_NODES 100000
#define N_EDGES 3200000
#define N_FEAT  500
#define HIDDEN  16
#define N_CLASS 40

typedef __attribute__((ext_vector_type(16))) __bf16       v16bf;
typedef __attribute__((ext_vector_type(2)))  __bf16       v2bf;
typedef __attribute__((ext_vector_type(8)))  float        v8f;
typedef __attribute__((ext_vector_type(8)))  unsigned int v8u;

#if defined(__has_builtin)
#  if __has_builtin(__builtin_amdgcn_cvt_pk_bf16_f32)
#    define HAVE_CVT_PK_BF16 1
#  endif
#endif

// f32 -> bf16 pack. Preferred: single-instruction VOP3P converts
// (v_fma_mixlo_bf16 / v_fma_mixhi_bf16, CDNA5 ISA §15.10 opc 62/63) —
// fma(v,1.0,0) == v, converted RNE into the selected half of the dword.
__device__ __forceinline__ unsigned int packbf(float lo, float hi) {
#ifdef HAVE_CVT_PK_BF16
  v2bf p = __builtin_amdgcn_cvt_pk_bf16_f32(lo, hi);
  return __builtin_bit_cast(unsigned int, p);
#else
  unsigned int d;
  asm("v_fma_mixlo_bf16 %0, %1, 1.0, 0\n\t"
      "v_fma_mixhi_bf16 %0, %2, 1.0, 0"
      : "=&v"(d)
      : "v"(lo), "v"(hi));
  return d;
#endif
}

__device__ __forceinline__ void atomAddF(float* p, float v) {
  unsafeAtomicAdd(p, v);   // hardware global_atomic_add_f32
}

// ---------------------------------------------------------------------------
// Pack [W1[0] | W1[1] | root1] (500x48, K padded to 512) into WMMA B-fragment
// layout: wbuf[((kt*3+nt)*32 + lane)*8 + j] ; lane: n=lane&15, khalf=lane>>4 ;
// dword j holds K pair ((j>>2)*16 + khalf*8 + 2*(j&3), +1) of K-tile kt.
// ---------------------------------------------------------------------------
__global__ __launch_bounds__(256) void prep_w1(const float* __restrict__ W1,
                                               const float* __restrict__ root1,
                                               unsigned int* __restrict__ wbuf) {
  int t = blockIdx.x * blockDim.x + threadIdx.x;
  if (t >= 16 * 3 * 32 * 8) return;
  int j    = t & 7;
  int lane = (t >> 3) & 31;
  int rest = t >> 8;
  int nt   = rest % 3;
  int kt   = rest / 3;
  int col  = nt * 16 + (lane & 15);
  int khalf = lane >> 4;
  float v[2];
#pragma unroll
  for (int s = 0; s < 2; ++s) {
    int k = kt * 32 + (j >> 2) * 16 + khalf * 8 + 2 * (j & 3) + s;
    float x = 0.f;
    if (k < N_FEAT) {
      if (col < 16)      x = W1[(size_t)k * HIDDEN + col];
      else if (col < 32) x = W1[(size_t)(N_FEAT + k) * HIDDEN + (col - 16)];
      else               x = root1[(size_t)k * HIDDEN + (col - 32)];
    }
    v[s] = x;
  }
  wbuf[t] = packbf(v[0], v[1]);
}

// Pack [W2[0] | W2[1] | root2] (16x120, K padded to 32, N padded to 128)
__global__ __launch_bounds__(256) void prep_w2(const float* __restrict__ W2,
                                               const float* __restrict__ root2,
                                               unsigned int* __restrict__ wbuf) {
  int t = blockIdx.x * blockDim.x + threadIdx.x;
  if (t >= 8 * 32 * 8) return;
  int j    = t & 7;
  int lane = (t >> 3) & 31;
  int nt   = t >> 8;
  int col  = nt * 16 + (lane & 15);
  int khalf = lane >> 4;
  float v[2];
#pragma unroll
  for (int s = 0; s < 2; ++s) {
    int k = (j >> 2) * 16 + khalf * 8 + 2 * (j & 3) + s;
    float x = 0.f;
    if (k < HIDDEN && col < 120) {
      if (col < 40)      x = W2[(size_t)k * N_CLASS + col];
      else if (col < 80) x = W2[(size_t)(HIDDEN + k) * N_CLASS + (col - 40)];
      else               x = root2[(size_t)k * N_CLASS + (col - 80)];
    }
    v[s] = x;
  }
  wbuf[t] = packbf(v[0], v[1]);
}

// ---------------------------------------------------------------------------
// GEMM1: xw[100000 x 48] = x[100000 x 500] @ Wcat[500 x 48], bf16 WMMA.
// W is staged into LDS once per block via gfx1250 async global->LDS copies
// (ASYNCcnt), then the K-loop reads B fragments with ds_load_b128.
// One wave per 16-row M-tile, 3 f32 accumulators, K loop of 16x32.
// ---------------------------------------------------------------------------
#define W1_DWORDS (16 * 3 * 32 * 8)   // 12288 dwords = 48 KB

__global__ __launch_bounds__(256) void gemm1(const float* __restrict__ x,
                                             const unsigned int* __restrict__ wbuf,
                                             float* __restrict__ xw) {
  __shared__ unsigned int sW[W1_DWORDS];

  // Async-stage packed W into LDS: 256 threads x 12 x b128 = 48 KB
  {
    unsigned lbase = (unsigned)(size_t)(&sW[0]);
    unsigned long long gbase = (unsigned long long)(size_t)wbuf;
#pragma unroll
    for (int p = 0; p < 12; ++p) {
      unsigned byteoff = (unsigned)((p * 256 + threadIdx.x) * 16);
      unsigned loff = lbase + byteoff;
      unsigned long long ga = gbase + byteoff;
      asm volatile("global_load_async_to_lds_b128 %0, %1, off"
                   :: "v"(loff), "v"(ga) : "memory");
    }
    asm volatile("s_wait_asynccnt 0x0" ::: "memory");
    __syncthreads();
  }

  int wid  = (int)((blockIdx.x * blockDim.x + threadIdx.x) >> 5);
  int lane = threadIdx.x & 31;
  if (wid < N_NODES / 16) {
    int m  = (wid << 4) + (lane & 15);
    int hf = lane >> 4;
    const float* xrow = x + (size_t)m * N_FEAT;

    v8f acc[3] = {};
    for (int kt = 0; kt < 16; ++kt) {
      int k0 = kt * 32 + 8 * hf;   // always <= 488: in range
      int k1 = k0 + 16;            // may cross 500 on the last tile
      v8u au;
      const float4* p0 = (const float4*)(xrow + k0);
      float4 a0 = p0[0], a1 = p0[1];
      au[0] = packbf(a0.x, a0.y); au[1] = packbf(a0.z, a0.w);
      au[2] = packbf(a1.x, a1.y); au[3] = packbf(a1.z, a1.w);
      // N_FEAT % 4 == 0 and loads are 4-aligned: a float4 at base < 500 is
      // fully in-bounds, so the tail needs only two conditional loads.
      float4 c0{0.f, 0.f, 0.f, 0.f}, c1{0.f, 0.f, 0.f, 0.f};
      if (k1 < N_FEAT)     c0 = *(const float4*)(xrow + k1);
      if (k1 + 4 < N_FEAT) c1 = *(const float4*)(xrow + k1 + 4);
      au[4] = packbf(c0.x, c0.y); au[5] = packbf(c0.z, c0.w);
      au[6] = packbf(c1.x, c1.y); au[7] = packbf(c1.z, c1.w);
      v16bf a = __builtin_bit_cast(v16bf, au);
#pragma unroll
      for (int nt = 0; nt < 3; ++nt) {
        const uint4* bp = (const uint4*)(sW + (size_t)(((kt * 3 + nt) * 32) + lane) * 8);
        uint4 q0 = bp[0], q1 = bp[1];
        v8u bu;
        bu[0] = q0.x; bu[1] = q0.y; bu[2] = q0.z; bu[3] = q0.w;
        bu[4] = q1.x; bu[5] = q1.y; bu[6] = q1.z; bu[7] = q1.w;
        v16bf b = __builtin_bit_cast(v16bf, bu);
        acc[nt] = __builtin_amdgcn_wmma_f32_16x16x32_bf16(
            false, a, false, b, (short)0, acc[nt], false, false);
      }
    }
    int col = lane & 15;
    int row0 = (wid << 4) + (hf << 3);
#pragma unroll
    for (int r = 0; r < 8; ++r) {
      float* o = xw + (size_t)(row0 + r) * 48;
      o[col]      = acc[0][r];
      o[16 + col] = acc[1][r];
      o[32 + col] = acc[2][r];
    }
  }
}

// ---------------------------------------------------------------------------
// GEMM2: hw[100000 x 120] = h[100000 x 16] @ Wcat2[16 x 120], single K tile.
// ---------------------------------------------------------------------------
__global__ __launch_bounds__(256) void gemm2(const float* __restrict__ h,
                                             const unsigned int* __restrict__ wbuf,
                                             float* __restrict__ hw) {
  int wid  = (int)((blockIdx.x * blockDim.x + threadIdx.x) >> 5);
  int lane = threadIdx.x & 31;
  if (wid >= N_NODES / 16) return;
  int m  = (wid << 4) + (lane & 15);
  int hf = lane >> 4;
  const float4* hrow = (const float4*)(h + (size_t)m * HIDDEN + 8 * hf);
  float4 a0 = hrow[0], a1 = hrow[1];
  v8u au;
  au[0] = packbf(a0.x, a0.y); au[1] = packbf(a0.z, a0.w);
  au[2] = packbf(a1.x, a1.y); au[3] = packbf(a1.z, a1.w);
  au[4] = 0u; au[5] = 0u; au[6] = 0u; au[7] = 0u;   // K=16..31 padding
  v16bf a = __builtin_bit_cast(v16bf, au);

  int col  = lane & 15;
  int row0 = (wid << 4) + (hf << 3);
#pragma unroll
  for (int nt = 0; nt < 8; ++nt) {
    const uint4* bp = (const uint4*)(wbuf + (size_t)(nt * 32 + lane) * 8);
    uint4 q0 = bp[0], q1 = bp[1];
    v8u bu;
    bu[0] = q0.x; bu[1] = q0.y; bu[2] = q0.z; bu[3] = q0.w;
    bu[4] = q1.x; bu[5] = q1.y; bu[6] = q1.z; bu[7] = q1.w;
    v16bf b = __builtin_bit_cast(v16bf, bu);
    v8f acc = {};
    acc = __builtin_amdgcn_wmma_f32_16x16x32_bf16(
        false, a, false, b, (short)0, acc, false, false);
    int c = nt * 16 + col;
    if (c < 120) {
#pragma unroll
      for (int r = 0; r < 8; ++r) hw[(size_t)(row0 + r) * 120 + c] = acc[r];
    }
  }
}

// ---------------------------------------------------------------------------
// Edge pass 1: one thread per (edge, feature<16); lerp + scatter-add (L2 atomics)
// ---------------------------------------------------------------------------
__global__ __launch_bounds__(256) void edge1(const int* __restrict__ ei,
                                             const float* __restrict__ attr,
                                             const float* __restrict__ xw,
                                             float* __restrict__ agg1,
                                             float* __restrict__ cnt) {
  long t = (long)blockIdx.x * blockDim.x + threadIdx.x;
  if (t >= (long)N_EDGES * 16) return;
  int e = (int)(t >> 4), f = (int)(t & 15);
  int s = ei[e], d = ei[N_EDGES + e];
  float u = attr[e];
  float msg = (1.f - u) * xw[(size_t)s * 48 + f] + u * xw[(size_t)s * 48 + 16 + f];
  atomAddF(&agg1[(size_t)d * 16 + f], msg);
  if (f == 0) atomAddF(&cnt[d], 1.f);
}

// Edge pass 2: one thread per (edge, class<40); counts reused from layer 1
__global__ __launch_bounds__(256) void edge2(const int* __restrict__ ei,
                                             const float* __restrict__ attr,
                                             const float* __restrict__ hw,
                                             float* __restrict__ agg2) {
  long t = (long)blockIdx.x * blockDim.x + threadIdx.x;
  if (t >= (long)N_EDGES * 40) return;
  int e = (int)(t / 40), f = (int)(t % 40);
  int s = ei[e], d = ei[N_EDGES + e];
  float u = attr[e];
  float msg = (1.f - u) * hw[(size_t)s * 120 + f] + u * hw[(size_t)s * 120 + 40 + f];
  atomAddF(&agg2[(size_t)d * 40 + f], msg);
}

// Node finish 1: mean + root term + bias, ELU -> h
__global__ __launch_bounds__(256) void finish1(const float* __restrict__ agg1,
                                               const float* __restrict__ cnt,
                                               const float* __restrict__ xw,
                                               const float* __restrict__ b1,
                                               float* __restrict__ h) {
  int t = blockIdx.x * blockDim.x + threadIdx.x;
  if (t >= N_NODES * HIDDEN) return;
  int n = t >> 4, f = t & 15;
  float c = fmaxf(cnt[n], 1.f);
  float a = agg1[t] / c + xw[(size_t)n * 48 + 32 + f] + b1[f];
  h[t] = a > 0.f ? a : (expf(a) - 1.f);
}

// Node finish 2: mean + root + bias, row-wise log_softmax -> out
__global__ __launch_bounds__(256) void finish2(const float* __restrict__ agg2,
                                               const float* __restrict__ cnt,
                                               const float* __restrict__ hw,
                                               const float* __restrict__ b2,
                                               float* __restrict__ out) {
  int n = blockIdx.x * blockDim.x + threadIdx.x;
  if (n >= N_NODES) return;
  float c = fmaxf(cnt[n], 1.f);
  float v[N_CLASS];
  float mx = -3.4e38f;
#pragma unroll
  for (int j = 0; j < N_CLASS; ++j) {
    float a = agg2[(size_t)n * 40 + j] / c + hw[(size_t)n * 120 + 80 + j] + b2[j];
    v[j] = a;
    mx = fmaxf(mx, a);
  }
  float se = 0.f;
#pragma unroll
  for (int j = 0; j < N_CLASS; ++j) se += expf(v[j] - mx);
  float lse = mx + logf(se);
#pragma unroll
  for (int j = 0; j < N_CLASS; ++j) out[(size_t)n * 40 + j] = v[j] - lse;
}

extern "C" void kernel_launch(void* const* d_in, const int* in_sizes, int n_in,
                              void* d_out, int out_size, void* d_ws, size_t ws_size,
                              hipStream_t stream) {
  const float* x     = (const float*)d_in[0];
  const float* attr  = (const float*)d_in[1];
  const float* W1    = (const float*)d_in[2];
  const float* root1 = (const float*)d_in[3];
  const float* b1    = (const float*)d_in[4];
  const float* W2    = (const float*)d_in[5];
  const float* root2 = (const float*)d_in[6];
  const float* b2    = (const float*)d_in[7];
  const int*   ei    = (const int*)d_in[8];
  float* out = (float*)d_out;

  float* ws   = (float*)d_ws;
  float* xw   = ws;                                 // 100000*48
  float* agg1 = xw   + (size_t)N_NODES * 48;        // 100000*16
  float* cnt  = agg1 + (size_t)N_NODES * 16;        // 100000   (contiguous w/ agg1)
  float* h    = cnt  + (size_t)N_NODES;             // 100000*16
  float* hw   = h    + (size_t)N_NODES * 16;        // 100000*120
  float* agg2 = hw   + (size_t)N_NODES * 120;       // 100000*40
  unsigned int* wbuf1 = (unsigned int*)(agg2 + (size_t)N_NODES * 40);  // 12288 dwords
  unsigned int* wbuf2 = wbuf1 + W1_DWORDS;                             // 2048 dwords

  // zero accumulators (agg1 + cnt are contiguous)
  hipMemsetAsync(agg1, 0, ((size_t)N_NODES * 16 + N_NODES) * sizeof(float), stream);
  hipMemsetAsync(agg2, 0, (size_t)N_NODES * 40 * sizeof(float), stream);

  prep_w1<<<48, 256, 0, stream>>>(W1, root1, wbuf1);
  prep_w2<<<8, 256, 0, stream>>>(W2, root2, wbuf2);

  int waves  = N_NODES / 16;                 // 6250 M-tiles
  int blocks = (waves * 32 + 255) / 256;     // 782 blocks, 8 waves each
  gemm1<<<blocks, 256, 0, stream>>>(x, wbuf1, xw);

  long e1 = (long)N_EDGES * 16;
  edge1<<<(int)((e1 + 255) / 256), 256, 0, stream>>>(ei, attr, xw, agg1, cnt);

  finish1<<<(N_NODES * HIDDEN + 255) / 256, 256, 0, stream>>>(agg1, cnt, xw, b1, h);

  gemm2<<<blocks, 256, 0, stream>>>(h, wbuf2, hw);

  long e2 = (long)N_EDGES * 40;
  edge2<<<(int)((e2 + 255) / 256), 256, 0, stream>>>(ei, attr, hw, agg2);

  finish2<<<(N_NODES + 255) / 256, 256, 0, stream>>>(agg2, cnt, hw, b2, out);
}
